// PositionFeaturizer_63101659512934
// MI455X (gfx1250) — matrix operand order repaired and checked by reading, not compiled, not run on hardware
//
#include <hip/hip_runtime.h>
#include <hip/hip_bf16.h>

// ---------------------------------------------------------------------------
// PositionFeaturizer for MI455X (gfx1250, wave32).
//   Phase 0: bf16 conversion of x; bf16 + transpose of Wq/Wk (one-time).
//   Phase 1: q = x@Wq+bq, k = x@Wk+bk via v_wmma_f32_16x16x32_bf16,
//            pure b128 loads in the K loop, 4 output tiles per wave.
//   Phase 2: per-row (16 contiguous edges) sparse attention + softmax +
//            1/dist reweight + position feature, wave-per-head.
// ---------------------------------------------------------------------------

#define NROWS 8192
#define DMODEL 512
#define NHEADS 8
#define HDIM 64
#define KEDGE 16
#define NEDGE (NROWS * KEDGE)

typedef __attribute__((ext_vector_type(16))) __bf16 v16bf;
typedef __attribute__((ext_vector_type(8)))  __bf16 v8bf;
typedef __attribute__((ext_vector_type(8)))  float  v8f;

__device__ __forceinline__ unsigned short f2bf_bits(float f) {
    union { float f; unsigned u; } c; c.f = f;
    unsigned r = c.u + 0x7FFFu + ((c.u >> 16) & 1u);   // round-to-nearest-even
    return (unsigned short)(r >> 16);
}

// ---- Phase 0a: x (fp32, [N,D]) -> bf16 row-major. 4 elems / thread. -------
__global__ __launch_bounds__(256)
void cvt_x_bf16(const float* __restrict__ in, unsigned short* __restrict__ out)
{
    const size_t i = ((size_t)blockIdx.x * 256 + threadIdx.x) * 4;
    // N*D = 4M, divisible by 4*256: no tail
    unsigned short o0 = f2bf_bits(in[i + 0]);
    unsigned short o1 = f2bf_bits(in[i + 1]);
    unsigned short o2 = f2bf_bits(in[i + 2]);
    unsigned short o3 = f2bf_bits(in[i + 3]);
    ushort2* p = (ushort2*)(out + i);
    p[0] = make_ushort2(o0, o1);
    p[1] = make_ushort2(o2, o3);
}

// ---- Phase 0b: Wq/Wk (fp32, [K,N]) -> bf16 transposed [N,K]. --------------
__global__ __launch_bounds__(256)
void cvt_w_bf16_t(const float* __restrict__ Wq, const float* __restrict__ Wk,
                  unsigned short* __restrict__ WqT, unsigned short* __restrict__ WkT)
{
    const int n = blockIdx.x;                 // output row (= W column)
    for (int k = threadIdx.x; k < DMODEL; k += 256) {
        WqT[(size_t)n * DMODEL + k] = f2bf_bits(Wq[(size_t)k * DMODEL + n]);
        WkT[(size_t)n * DMODEL + k] = f2bf_bits(Wk[(size_t)k * DMODEL + n]);
    }
}

// ---- Phase 1: WMMA GEMM. One wave -> 16 rows x 64 cols (4 tiles). ---------
// grid = (NROWS/16, 8 * 2): y&7 selects 64-wide n-group, y>>3 selects q|k.
__global__ __launch_bounds__(32)
void proj_gemm_wmma(const unsigned short* __restrict__ xb,   // bf16 [N,D]
                    const unsigned short* __restrict__ WqT,  // bf16 [D(n),D(k)]
                    const unsigned short* __restrict__ WkT,
                    const float* __restrict__ bq, const float* __restrict__ bk,
                    float* __restrict__ q_out, float* __restrict__ k_out)
{
    const int lane  = threadIdx.x;            // 0..31
    const int m0    = blockIdx.x * 16;
    const int nbase = (blockIdx.y & 7) * 64;  // 4 tiles of 16
    const int which = blockIdx.y >> 3;        // 0 = q, 1 = k

    const unsigned short* __restrict__ WT = which ? WkT : WqT;
    const float* __restrict__ bias        = which ? bk  : bq;
    float* __restrict__ outp              = which ? k_out : q_out;

    const int half = lane >> 4;               // 0 | 1
    const int l16  = lane & 15;

    // Per-lane fragment base pointers (ISA 7.12.2 16-bit layouts):
    //   A: row m0+l16; elems 0..7 = K kk+half*8.., elems 8..15 = K kk+16+half*8..
    //   B: col n+l16;  elems 0..7 = K kk+half*8.., elems 8..15 = K kk+16+half*8..
    const unsigned short* arow = xb + (size_t)(m0 + l16) * DMODEL + half * 8;

    v8f acc[4] = {};

    for (int kk = 0; kk < DMODEL; kk += 32) {
        const v8bf alo = *(const v8bf*)(arow + kk);        // 16B aligned
        const v8bf ahi = *(const v8bf*)(arow + kk + 16);
        const v16bf a = __builtin_shufflevector(alo, ahi,
                            0,1,2,3,4,5,6,7,8,9,10,11,12,13,14,15);
        __builtin_prefetch(arow + kk + 32, 0, 1);          // next A fragment

        #pragma unroll
        for (int t = 0; t < 4; ++t) {
            const unsigned short* brow =
                WT + (size_t)(nbase + t * 16 + l16) * DMODEL + kk + half * 8;
            const v8bf blo = *(const v8bf*)(brow);
            const v8bf bhi = *(const v8bf*)(brow + 16);
            const v16bf bm = __builtin_shufflevector(blo, bhi,
                                 0,1,2,3,4,5,6,7,8,9,10,11,12,13,14,15);
            acc[t] = __builtin_amdgcn_wmma_f32_16x16x32_bf16(
                         false, a, false, bm, (short)0, acc[t], false, false);
        }
    }

    // C/D layout: VGPR r -> row (r + 8*half), col l16 (within tile)
    #pragma unroll
    for (int t = 0; t < 4; ++t) {
        const int n = nbase + t * 16 + l16;
        const float bv = bias[n];
        #pragma unroll
        for (int r = 0; r < 8; ++r) {
            outp[(size_t)(m0 + half * 8 + r) * DMODEL + n] = acc[t][r] + bv;
        }
    }
}

// ---- Phase 2: one block per row; blockDim = (32, 8): wave = one head. -----
__global__ __launch_bounds__(256)
void edge_attn_feat(const float* __restrict__ q,
                    const float* __restrict__ k,
                    const int*   __restrict__ src_index,
                    const float* __restrict__ att_bias,   // [H, E]
                    const float* __restrict__ dist,       // [E]
                    const float* __restrict__ pos,        // [N, 3]
                    const float* __restrict__ src_pos,    // [S, 3]
                    const int*   __restrict__ org_to_src, // [S]
                    float* __restrict__ out)              // [N, H*3]
{
    const int row  = blockIdx.x;
    const int h    = threadIdx.y;      // 0..7
    const int lane = threadIdx.x;      // 0..31
    const int e0   = row * KEDGE;

    const float* qh = q + (size_t)row * DMODEL + h * HDIM;
    const float q0 = qh[lane];
    const float q1 = qh[lane + 32];

    float logits[KEDGE];
    int   srcs[KEDGE];

    #pragma unroll
    for (int e = 0; e < KEDGE; ++e) {
        const int s    = src_index[e0 + e];
        srcs[e]        = s;
        const int srow = org_to_src[s];
        const float* kh = k + (size_t)srow * DMODEL + h * HDIM;
        float p = q0 * kh[lane] + q1 * kh[lane + 32];
        p += __shfl_xor(p, 16);
        p += __shfl_xor(p,  8);
        p += __shfl_xor(p,  4);
        p += __shfl_xor(p,  2);
        p += __shfl_xor(p,  1);
        logits[e] = p * 0.125f   // 1/sqrt(64)
                  + att_bias[(size_t)h * NEDGE + e0 + e];
    }

    float m = logits[0];
    #pragma unroll
    for (int e = 1; e < KEDGE; ++e) m = fmaxf(m, logits[e]);
    float pv[KEDGE];
    float z = 0.f;
    #pragma unroll
    for (int e = 0; e < KEDGE; ++e) { pv[e] = __expf(logits[e] - m); z += pv[e]; }
    const float invz = 1.0f / z;

    float d0 = 0.f, d1 = 0.f, d2 = 0.f, asum = 0.f;
    #pragma unroll
    for (int e = 0; e < KEDGE; ++e) {
        const float dd = dist[e0 + e];
        const float w  = (dd == 0.f) ? 0.f : (pv[e] * invz / dd);
        const int   s  = srcs[e];
        d0   += w * src_pos[(size_t)s * 3 + 0];
        d1   += w * src_pos[(size_t)s * 3 + 1];
        d2   += w * src_pos[(size_t)s * 3 + 2];
        asum += w;
    }

    if (lane == 0) {
        const float p0 = pos[(size_t)row * 3 + 0];
        const float p1 = pos[(size_t)row * 3 + 1];
        const float p2 = pos[(size_t)row * 3 + 2];
        float* o = out + (size_t)row * (NHEADS * 3) + h * 3;
        o[0] = d0 - asum * p0;
        o[1] = d1 - asum * p1;
        o[2] = d2 - asum * p2;
    }
}

extern "C" void kernel_launch(void* const* d_in, const int* in_sizes, int n_in,
                              void* d_out, int out_size, void* d_ws, size_t ws_size,
                              hipStream_t stream) {
    const float* x          = (const float*)d_in[0];
    // d_in[1] = row_index: unused (edges are contiguous, 16 per row)
    const int*   src_index  = (const int*)  d_in[2];
    const float* att_bias   = (const float*)d_in[3];
    const float* dist       = (const float*)d_in[4];
    const float* pos        = (const float*)d_in[5];
    const float* src_pos    = (const float*)d_in[6];
    const int*   org_to_src = (const int*)  d_in[7];
    const float* Wq         = (const float*)d_in[8];
    const float* bq         = (const float*)d_in[9];
    const float* Wk         = (const float*)d_in[10];
    const float* bk         = (const float*)d_in[11];
    float* out = (float*)d_out;

    // workspace carve-up
    float*          q_ws = (float*)d_ws;                          // 16 MB
    float*          k_ws = q_ws + (size_t)NROWS * DMODEL;         // 16 MB
    unsigned short* xb   = (unsigned short*)(k_ws + (size_t)NROWS * DMODEL); // 8 MB
    unsigned short* WqT  = xb  + (size_t)NROWS * DMODEL;          // 0.5 MB
    unsigned short* WkT  = WqT + (size_t)DMODEL * DMODEL;         // 0.5 MB

    // Phase 0: conversions
    cvt_x_bf16<<<(NROWS * DMODEL) / (256 * 4), 256, 0, stream>>>(x, xb);
    cvt_w_bf16_t<<<DMODEL, 256, 0, stream>>>(Wq, Wk, WqT, WkT);

    // Phase 1: WMMA projections
    dim3 g1(NROWS / 16, 2 * (DMODEL / 64));   // 512 x 16
    proj_gemm_wmma<<<g1, 32, 0, stream>>>(xb, WqT, WkT, bq, bk, q_ws, k_ws);

    // Phase 2: edge attention + features
    dim3 b2(32, NHEADS);
    edge_attn_feat<<<NROWS, b2, 0, stream>>>(q_ws, k_ws, src_index, att_bias,
                                             dist, pos, src_pos, org_to_src, out);
}
